// MoEFFN_81647328297468
// MI455X (gfx1250) — matrix operand order repaired and checked by reading, not compiled, not run on hardware
//
#include <hip/hip_runtime.h>

// ---------------------------------------------------------------------------
// MoE FFN for MI455X (gfx1250, wave32, WMMA bf16 16x16x32)
// ---------------------------------------------------------------------------

typedef __attribute__((ext_vector_type(16))) __bf16 v16bf;
typedef __attribute__((ext_vector_type(8)))  __bf16 v8bf;
typedef __attribute__((ext_vector_type(8)))  float  v8f;

#define T_TOTAL 16384   // B*S tokens
#define DMODEL  1024
#define DEXP    128
#define NEXP    8
#define HSTRIDE 1056    // 1024 + 32 pad (bf16 elems) -> 16B-aligned rows, fewer bank conflicts

__device__ __forceinline__ unsigned short f2bf(float f) {
    union { float f; unsigned u; } v; v.f = f;
    unsigned r = v.u + 0x7fffu + ((v.u >> 16) & 1u);   // round-to-nearest-even
    return (unsigned short)(r >> 16);
}

// ---------------------------------------------------------------------------
// Kernel 1: fp32 -> bf16 conversion of W1 and W2 (both are E*DE*D = 1M elems)
// ---------------------------------------------------------------------------
__global__ void moe_wconv_kernel(const float* __restrict__ W1,
                                 const float* __restrict__ W2,
                                 unsigned short* __restrict__ W1bf,
                                 unsigned short* __restrict__ W2bf, int n) {
    for (int i = blockIdx.x * blockDim.x + threadIdx.x; i < n;
         i += gridDim.x * blockDim.x) {
        W1bf[i] = f2bf(W1[i]);
        W2bf[i] = f2bf(W2[i]);
    }
}

// ---------------------------------------------------------------------------
// Kernel 2: gate (softmax -> top-2 -> renorm) + bf16 copy of x.
// One wave32 per token; 8 waves per block.
// ---------------------------------------------------------------------------
__global__ __launch_bounds__(256)
void moe_gate_kernel(const float* __restrict__ x, const float* __restrict__ Wg,
                     unsigned short* __restrict__ xbf, float* __restrict__ gate) {
    const int lane = threadIdx.x & 31;
    const int wave = threadIdx.x >> 5;
    const int t    = blockIdx.x * 8 + wave;
    const float* xr = x + (size_t)t * DMODEL;

    float acc[NEXP];
#pragma unroll
    for (int e = 0; e < NEXP; ++e) acc[e] = 0.f;

    for (int d = lane; d < DMODEL; d += 32) {
        float xv = xr[d];
        xbf[(size_t)t * DMODEL + d] = f2bf(xv);
#pragma unroll
        for (int e = 0; e < NEXP; ++e) acc[e] += xv * Wg[e * DMODEL + d];
    }
    // wave32 butterfly reduction: every lane ends with the full logits
#pragma unroll
    for (int e = 0; e < NEXP; ++e) {
#pragma unroll
        for (int off = 16; off > 0; off >>= 1)
            acc[e] += __shfl_xor(acc[e], off, 32);
    }
    // softmax over 8 experts
    float mx = acc[0];
#pragma unroll
    for (int e = 1; e < NEXP; ++e) mx = fmaxf(mx, acc[e]);
    float p[NEXP], s = 0.f;
#pragma unroll
    for (int e = 0; e < NEXP; ++e) { p[e] = __expf(acc[e] - mx); s += p[e]; }
    float inv = 1.f / s;
#pragma unroll
    for (int e = 0; e < NEXP; ++e) p[e] *= inv;
    // top-2
    int i1 = 0; float w1 = p[0];
#pragma unroll
    for (int e = 1; e < NEXP; ++e) if (p[e] > w1) { w1 = p[e]; i1 = e; }
    int i2 = -1; float w2 = -1.f;
#pragma unroll
    for (int e = 0; e < NEXP; ++e) if (e != i1 && p[e] > w2) { w2 = p[e]; i2 = e; }
    const float rn = 1.f / (w1 + w2);
    if (lane < NEXP) {
        float g = (lane == i1) ? w1 * rn : (lane == i2) ? w2 * rn : 0.f;
        gate[t * NEXP + lane] = g;
    }
}

// ---------------------------------------------------------------------------
// Kernel 3: expert FFN. One wave32 per 16-token tile.
// fc1: per expert, 16x128 = 8 n-tiles, K=1024 in 32 steps of WMMA bf16.
// Epilogue: +b1, relu, *gate -> bf16 into LDS as concatenated H [16 x 1024].
// fc2: y[16 x 1024] = H @ W2cat^T (+ gate@b2) as one K=1024 WMMA GEMM.
// ---------------------------------------------------------------------------
__global__ __launch_bounds__(32)
void moe_ffn_kernel(const unsigned short* __restrict__ xbf,
                    const unsigned short* __restrict__ W1bf,
                    const unsigned short* __restrict__ W2bf,
                    const float* __restrict__ b1,
                    const float* __restrict__ b2,
                    const float* __restrict__ gate,
                    float* __restrict__ y) {
    __shared__ __align__(16) unsigned short hbuf[16 * HSTRIDE];

    const int lane = threadIdx.x;      // 0..31
    const int half = lane >> 4;        // K-half selector for A/B fragments
    const int mr   = lane & 15;        // M (A, rows) / N (B, cols) index
    const int t0   = blockIdx.x * 16;  // first token of this tile

    // gate weights mapped to the C/D accumulator layout:
    // element r of a v8f accumulator is row m = r + 8*half
    float gall[NEXP][8];
#pragma unroll
    for (int e = 0; e < NEXP; ++e)
#pragma unroll
        for (int r = 0; r < 8; ++r)
            gall[e][r] = gate[(size_t)(t0 + r + 8 * half) * NEXP + e];

    const unsigned short* Ar = xbf + (size_t)(t0 + mr) * DMODEL;

    // ---------------- fc1 for each expert ----------------
    for (int e = 0; e < NEXP; ++e) {
        v8f acc[8];
#pragma unroll
        for (int nt = 0; nt < 8; ++nt) acc[nt] = (v8f)0.f;

        for (int kt = 0; kt < DMODEL / 32; ++kt) {
            const int k0 = kt * 32;
            __builtin_prefetch(Ar + k0 + 64, 0, 1);   // global_prefetch_b8
            // A fragment: 16 contiguous k per lane half (k = 16*half + s)
            v8bf alo = *(const v8bf*)(Ar + k0 + 16 * half);
            v8bf ahi = *(const v8bf*)(Ar + k0 + 16 * half + 8);
            v16bf a = __builtin_shufflevector(alo, ahi,
                0, 1, 2, 3, 4, 5, 6, 7, 8, 9, 10, 11, 12, 13, 14, 15);
#pragma unroll
            for (int nt = 0; nt < 8; ++nt) {
                // B[k][n] = W1[e][n][k]; 16 contiguous bf16 along k
                const unsigned short* Bp =
                    W1bf + (size_t)(e * DEXP + nt * 16 + mr) * DMODEL
                         + k0 + 16 * half;
                v16bf b = *(const v16bf*)Bp;
                acc[nt] = __builtin_amdgcn_wmma_f32_16x16x32_bf16(
                    false, a, false, b, (short)0, acc[nt], false, false);
            }
        }
        // epilogue: bias + relu + gate scale -> LDS (concatenated over experts)
#pragma unroll
        for (int nt = 0; nt < 8; ++nt) {
            const float bias = b1[e * DEXP + nt * 16 + mr];
#pragma unroll
            for (int r = 0; r < 8; ++r) {
                float v = acc[nt][r] + bias;
                v = v > 0.f ? v : 0.f;
                v *= gall[e][r];
                hbuf[(r + 8 * half) * HSTRIDE + e * DEXP + nt * 16 + mr] = f2bf(v);
            }
        }
    }

    // ---------------- fc2: y = H @ W2cat^T + gate@b2 ----------------
    for (int nt = 0; nt < DMODEL / 16; ++nt) {
        const int n = nt * 16 + mr;
        v8f acc = (v8f)0.f;
#pragma unroll
        for (int e = 0; e < NEXP; ++e) {
            const float bv = b2[e * DMODEL + n];
#pragma unroll
            for (int r = 0; r < 8; ++r) acc[r] += gall[e][r] * bv;
        }
        for (int kt = 0; kt < (NEXP * DEXP) / 32; ++kt) {
            const int k0 = kt * 32;
            // A fragment from LDS H tile
            v8bf alo = *(const v8bf*)(&hbuf[mr * HSTRIDE + k0 + 16 * half]);
            v8bf ahi = *(const v8bf*)(&hbuf[mr * HSTRIDE + k0 + 16 * half + 8]);
            v16bf a = __builtin_shufflevector(alo, ahi,
                0, 1, 2, 3, 4, 5, 6, 7, 8, 9, 10, 11, 12, 13, 14, 15);
            // B[kcat][n] = W2[e][n][kde]; each 32-wide tile stays in one expert
            const int ek = k0 >> 7;
            const int kd = (k0 & 127) + 16 * half;
            const unsigned short* Bp =
                W2bf + (size_t)(ek * DMODEL + n) * DEXP + kd;
            v16bf b = *(const v16bf*)Bp;
            acc = __builtin_amdgcn_wmma_f32_16x16x32_bf16(
                false, a, false, b, (short)0, acc, false, false);
        }
#pragma unroll
        for (int r = 0; r < 8; ++r)
            y[(size_t)(t0 + r + 8 * half) * DMODEL + n] = acc[r];
    }
}

// ---------------------------------------------------------------------------
// Host-side launch
// ---------------------------------------------------------------------------
extern "C" void kernel_launch(void* const* d_in, const int* in_sizes, int n_in,
                              void* d_out, int out_size, void* d_ws, size_t ws_size,
                              hipStream_t stream) {
    const float* x  = (const float*)d_in[0];   // [4,4096,1024]
    const float* Wg = (const float*)d_in[1];   // [8,1024]
    const float* W1 = (const float*)d_in[2];   // [8,128,1024]
    const float* b1 = (const float*)d_in[3];   // [8,128]
    const float* W2 = (const float*)d_in[4];   // [8,1024,128]
    const float* b2 = (const float*)d_in[5];   // [8,1024]
    float* y = (float*)d_out;                  // [4,4096,1024]

    // workspace layout (bytes)
    char* ws = (char*)d_ws;
    unsigned short* xbf  = (unsigned short*)(ws);                        // 33,554,432 B
    unsigned short* W1bf = (unsigned short*)(ws + 33554432);             //  2,097,152 B
    unsigned short* W2bf = (unsigned short*)(ws + 35651584);             //  2,097,152 B
    float*          gate = (float*)        (ws + 37748736);              //    524,288 B

    const int NW = NEXP * DEXP * DMODEL;  // 1,048,576 weight elems each
    moe_wconv_kernel<<<1024, 256, 0, stream>>>(W1, W2, W1bf, W2bf, NW);
    moe_gate_kernel<<<T_TOTAL / 8, 256, 0, stream>>>(x, Wg, xbf, gate);
    moe_ffn_kernel<<<T_TOTAL / 16, 32, 0, stream>>>(xbf, W1bf, W2bf, b1, b2, gate, y);
}
